// Deep_Lagrangian_Network_20968030339112
// MI455X (gfx1250) — matrix operand order repaired and checked by reading, not compiled, not run on hardware
//
#include <hip/hip_runtime.h>
#include <hip/hip_bf16.h>
#include <math.h>

// ---------------------------------------------------------------------------
// DeLaN (Deep Lagrangian Network) fused kernel for gfx1250 (MI455X, wave32).
//
// Per sample, 8 "streams" (primal + 7 JVP tangents) share the 256x256 hidden
// layers -> effective GEMM M=8*B, K=N=256, done with v_wmma_f32_16x16x32_f16
// (f16 in, f32 accumulate).  M=16 tile = 2 samples x 8 streams per wave; the
// WMMA C-layout places each sample's 8 streams in a single lane, so the
// softplus/sigmoid JVP gating is lane-local VALU work.
// ---------------------------------------------------------------------------

typedef __attribute__((ext_vector_type(16))) _Float16 v16h;
typedef __attribute__((ext_vector_type(8)))  float    v8f;

#define GLOBAL_AS __attribute__((address_space(1)))

#define N_DOF    7
#define WIDTH    256
#define BATCH    65536
#define N_TRIL   21
#define L_OFFSET 1e-3f

#define WAVES    4          // waves per workgroup
#define SPB      (WAVES*2)  // samples per workgroup (2 per wave)
#define ABUF_H   4096       // halfs per swizzled A buffer: 8 kt * 32 lanes * 16 elems
#define HB_COLS  48         // head columns padded (7 g + 7 Ld + 21 Lt + pad)

// workspace layout (halfs)
#define WS_W1   0
#define WS_W2   65536
#define WS_WH   131072

// ---------------------------------------------------------------------------
// device helpers
// ---------------------------------------------------------------------------

// Force a global (addrspace 1) load: inttoptr into AS(1) decouples the weight
// stream onto LOADcnt (global_load) instead of flat_load's LOADcnt+DScnt.
__device__ __forceinline__ v16h gload_v16h(const _Float16* p) {
    return *(GLOBAL_AS const v16h*)(unsigned long long)p;
}
__device__ __forceinline__ float gload_f32(const float* p) {
    return *(GLOBAL_AS const float*)(unsigned long long)p;
}

// branchless fast softplus / sigmoid (no EXEC-mask churn around the WMMA core)
__device__ __forceinline__ float softplus_f(float z) {
    const float e = __expf(-fabsf(z));
    return fmaxf(z, 0.f) + __logf(1.f + e);
}
__device__ __forceinline__ float sigmoid_f(float z) {
    return __fdividef(1.f, 1.f + __expf(-z));
}

// Store value into LDS buffer pre-swizzled as a 16-bit A-matrix (16x32)
// fragment sequence.  (m, col) -> (kt, lane, elem) in this code's contiguous k order (same as the B swizzle):
//   lane m (0-15) holds K in {0..15}; lane m+16 holds K in {16..31}
__device__ __forceinline__ void aswz_store(_Float16* buf, int m, int col, float v) {
    int kt   = col >> 5;
    int kl   = col & 31;
    int lane = m + ((kl & 16) ? 16 : 0);
    int e    = kl & 15;
    buf[((kt << 5) + lane) * 16 + e] = (_Float16)v;
}

// ---------------------------------------------------------------------------
// prep kernels: convert f32 weights to f16 in pre-swizzled B-fragment layout.
// B 32x16 (KxN) fragment, lane n = l&15, K = kt*32 + (l>>4)*16 + e.
// dst element index d = ((nt*8 + kt)*32 + lane)*16 + e  (contiguous per lane)
// ---------------------------------------------------------------------------
__global__ void delan_swz_w256(const float* __restrict__ src,
                               _Float16* __restrict__ dst, int total) {
    int d = blockIdx.x * blockDim.x + threadIdx.x;
    if (d >= total) return;
    int tile = d >> 9;          // / 512
    int r    = d & 511;
    int ln   = r >> 4;
    int e    = r & 15;
    int nt   = tile >> 3;
    int kt   = tile & 7;
    int n    = nt * 16 + (ln & 15);
    int K    = kt * 32 + ((ln >> 4) << 4) + e;
    dst[d] = (_Float16)src[K * WIDTH + n];
}

__global__ void delan_swz_heads(const float* __restrict__ Wg,
                                const float* __restrict__ Wd,
                                const float* __restrict__ Wt,
                                _Float16* __restrict__ dst, int total) {
    int d = blockIdx.x * blockDim.x + threadIdx.x;
    if (d >= total) return;
    int tile = d >> 9;
    int r    = d & 511;
    int ln   = r >> 4;
    int e    = r & 15;
    int nt   = tile >> 3;
    int kt   = tile & 7;
    int n    = nt * 16 + (ln & 15);          // 0..47
    int K    = kt * 32 + ((ln >> 4) << 4) + e;
    float v = 0.f;
    if      (n < 7)  v = Wg[K * N_DOF  + n];
    else if (n < 14) v = Wd[K * N_DOF  + (n - 7)];
    else if (n < 35) v = Wt[K * N_TRIL + (n - 14)];
    dst[d] = (_Float16)v;
}

// ---------------------------------------------------------------------------
// main fused kernel
// ---------------------------------------------------------------------------
__launch_bounds__(WAVES * 32)
__global__ void delan_main(const float* __restrict__ q,
                           const float* __restrict__ qd,
                           const float* __restrict__ qdd,
                           const float* __restrict__ W0,
                           const float* __restrict__ b0,
                           const float* __restrict__ b1,
                           const float* __restrict__ b2,
                           const float* __restrict__ bg,
                           const float* __restrict__ bd,
                           const float* __restrict__ bt,
                           const _Float16* __restrict__ w1s,
                           const _Float16* __restrict__ w2s,
                           const _Float16* __restrict__ whs,
                           float* __restrict__ out) {
    extern __shared__ __align__(16) char smem_raw[];
    _Float16* aswz    = (_Float16*)smem_raw;                         // WAVES*2*ABUF_H halfs
    float*    headbuf = (float*)(smem_raw + WAVES * 2 * ABUF_H * 2); // SPB*8*HB_COLS floats

    const int tid   = threadIdx.x;
    const int wave  = tid >> 5;
    const int lane  = tid & 31;
    const int mbase = (lane >> 4) << 3;   // 0 or 8: which sample-in-wave this lane carries
    const int wgStart = blockIdx.x * SPB;

    _Float16* bufA = aswz + (wave * 2 + 0) * ABUF_H;
    _Float16* bufB = aswz + (wave * 2 + 1) * ABUF_H;

    // ---------------- layer 0 (K=7, scalar) + JVP seeds --------------------
    // primal: h1 = softplus(q@W0 + b0); tangent c: sigmoid(z1) * W0[c, :]
    for (int samp = 0; samp < 2; ++samp) {
        const int s = wgStart + wave * 2 + samp;
        float qv[N_DOF];
#pragma unroll
        for (int k = 0; k < N_DOF; ++k) qv[k] = gload_f32(q + s * N_DOF + k);
#pragma unroll 1
        for (int j = 0; j < 8; ++j) {
            const int col = j * 32 + lane;
            float w0c[N_DOF];
#pragma unroll
            for (int k = 0; k < N_DOF; ++k) w0c[k] = gload_f32(W0 + k * WIDTH + col);
            float z = gload_f32(b0 + col);
#pragma unroll
            for (int k = 0; k < N_DOF; ++k) z += qv[k] * w0c[k];
            const float sg = sigmoid_f(z);
            aswz_store(bufA, samp * 8 + 0, col, softplus_f(z));
#pragma unroll
            for (int c = 0; c < N_DOF; ++c)
                aswz_store(bufA, samp * 8 + 1 + c, col, sg * w0c[c]);
        }
    }

    // ---------------- hidden layers 1 & 2 (WMMA) ---------------------------
    const _Float16* Ws[2] = { w1s, w2s };
    const float*    bs[2] = { b1, b2 };
    _Float16* rd = bufA;
    _Float16* wr = bufB;

#pragma unroll 1
    for (int l = 0; l < 2; ++l) {
        v16h Af[8];
#pragma unroll
        for (int kt = 0; kt < 8; ++kt)
            Af[kt] = *(const v16h*)(rd + ((kt << 5) + lane) * 16);

#pragma unroll 1
        for (int nt = 0; nt < 16; ++nt) {
            v8f acc = {};
            const _Float16* wb = Ws[l] + ((nt * 8) << 9) + lane * 16;
            __builtin_prefetch(Ws[l] + (((nt + 1) & 15) * 8 << 9), 0, 1);
#pragma unroll
            for (int kt = 0; kt < 8; ++kt) {
                v16h Bf = gload_v16h(wb + (kt << 9));
                acc = __builtin_amdgcn_wmma_f32_16x16x32_f16(
                          false, Af[kt], false, Bf, (short)0, acc, false, false);
            }
            const int col = nt * 16 + (lane & 15);
            const float z0 = acc[0] + gload_f32(bs[l] + col);  // bias: primal stream only
            const float sg = sigmoid_f(z0);
            aswz_store(wr, mbase + 0, col, softplus_f(z0));
#pragma unroll
            for (int r = 1; r < 8; ++r)
                aswz_store(wr, mbase + r, col, sg * acc[r]);
        }
        _Float16* t = rd; rd = wr; wr = t;
    }

    // ---------------- head GEMM (N=48 padded, WMMA) ------------------------
    {
        v16h Af[8];
#pragma unroll
        for (int kt = 0; kt < 8; ++kt)
            Af[kt] = *(const v16h*)(rd + ((kt << 5) + lane) * 16);

        const int sLocal = wave * 2 + (lane >> 4);
#pragma unroll 1
        for (int nt = 0; nt < 3; ++nt) {
            v8f acc = {};
            const _Float16* wb = whs + ((nt * 8) << 9) + lane * 16;
#pragma unroll
            for (int kt = 0; kt < 8; ++kt) {
                v16h Bf = gload_v16h(wb + (kt << 9));
                acc = __builtin_amdgcn_wmma_f32_16x16x32_f16(
                          false, Af[kt], false, Bf, (short)0, acc, false, false);
            }
            const int col = nt * 16 + (lane & 15);
            float bb = 0.f;
            if      (col < 7)  bb = gload_f32(bg + col);
            else if (col < 14) bb = gload_f32(bd + col - 7);
            else if (col < 35) bb = gload_f32(bt + col - 14);
#pragma unroll
            for (int r = 0; r < 8; ++r)
                headbuf[(sLocal * 8 + r) * HB_COLS + col] = acc[r] + (r == 0 ? bb : 0.f);
        }
    }

    __syncthreads();

    // ---------------- per-sample 7x7 physics (scalar epilogue) -------------
    if (tid < SPB) {
        const int s = wgStart + tid;
        const float* hb = headbuf + tid * 8 * HB_COLS;

        float qdv[N_DOF], qddv[N_DOF];
#pragma unroll
        for (int i = 0; i < N_DOF; ++i) {
            qdv[i]  = gload_f32(qd  + s * N_DOF + i);
            qddv[i] = gload_f32(qdd + s * N_DOF + i);
        }

        float gv[N_DOF], Ldiag[N_DOF], msk[N_DOF], Ltl[N_TRIL];
#pragma unroll
        for (int i = 0; i < N_DOF; ++i) {
            gv[i] = hb[i];
            const float raw = hb[7 + i];
            msk[i]   = (raw > 0.f) ? 1.f : 0.f;
            Ldiag[i] = fmaxf(raw, 0.f) + L_OFFSET;
        }
#pragma unroll
        for (int t = 0; t < N_TRIL; ++t) Ltl[t] = hb[14 + t];

        // time derivatives of L entries: sum over tangents c weighted by qd[c]
        float ld_dt[N_DOF], lt_dt[N_TRIL];
#pragma unroll
        for (int i = 0; i < N_DOF; ++i) {
            float a = 0.f;
#pragma unroll
            for (int c = 0; c < N_DOF; ++c) a += hb[(1 + c) * HB_COLS + 7 + i] * qdv[c];
            ld_dt[i] = msk[i] * a;
        }
#pragma unroll
        for (int t = 0; t < N_TRIL; ++t) {
            float a = 0.f;
#pragma unroll
            for (int c = 0; c < N_DOF; ++c) a += hb[(1 + c) * HB_COLS + 14 + t] * qdv[c];
            lt_dt[t] = a;
        }

        float L[N_DOF][N_DOF], Ldt[N_DOF][N_DOF];
#pragma unroll
        for (int i = 0; i < N_DOF; ++i)
#pragma unroll
            for (int j = 0; j < N_DOF; ++j) {
                if (i == j)     { L[i][j] = Ldiag[i];              Ldt[i][j] = ld_dt[i]; }
                else if (i > j) { const int t = i*(i-1)/2 + j;
                                  L[i][j] = Ltl[t];                Ldt[i][j] = lt_dt[t]; }
                else            { L[i][j] = 0.f;                   Ldt[i][j] = 0.f; }
            }

        // H = L L^T ; Hdt = L Ldt^T + Ldt L^T  (== H_dq contracted with qd)
        float H[N_DOF][N_DOF], Hdt[N_DOF][N_DOF];
#pragma unroll
        for (int i = 0; i < N_DOF; ++i)
#pragma unroll
            for (int k = 0; k < N_DOF; ++k) {
                float h = 0.f, hd = 0.f;
#pragma unroll
                for (int j = 0; j < N_DOF; ++j) {
                    h  += L[i][j] * L[k][j];
                    hd += L[i][j] * Ldt[k][j] + Ldt[i][j] * L[k][j];
                }
                H[i][k] = h; Hdt[i][k] = hd;
            }

        // c = Hdt qd - 0.5 * (qd^T Hdt) ; tau = H qdd + c + g
        float* outTau = out;
        float* outH   = out + (size_t)BATCH * 7;
        float* outC   = out + (size_t)BATCH * 56;
        float* outG   = out + (size_t)BATCH * 63;
#pragma unroll
        for (int i = 0; i < N_DOF; ++i) {
            float cv = 0.f, quad = 0.f, tv = 0.f;
#pragma unroll
            for (int j = 0; j < N_DOF; ++j) {
                cv   += Hdt[i][j] * qdv[j];
                quad += qdv[j] * Hdt[j][i];
                tv   += H[i][j] * qddv[j];
                outH[(size_t)s * 49 + i * 7 + j] = H[i][j];
            }
            cv -= 0.5f * quad;
            outC[(size_t)s * N_DOF + i]   = cv;
            outG[(size_t)s * N_DOF + i]   = gv[i];
            outTau[(size_t)s * N_DOF + i] = tv + cv + gv[i];
        }
    }
}

// ---------------------------------------------------------------------------
// launch
// ---------------------------------------------------------------------------
extern "C" void kernel_launch(void* const* d_in, const int* in_sizes, int n_in,
                              void* d_out, int out_size, void* d_ws, size_t ws_size,
                              hipStream_t stream) {
    (void)in_sizes; (void)n_in; (void)out_size; (void)ws_size;
    // setup_inputs dict order: q, qd, qdd, then params: W0,b0,W1,b1,W2,b2,Wg,bg,Wd,bd,Wt,bt
    const float* q   = (const float*)d_in[0];
    const float* qd  = (const float*)d_in[1];
    const float* qdd = (const float*)d_in[2];
    const float* W0  = (const float*)d_in[3];
    const float* b0  = (const float*)d_in[4];
    const float* W1  = (const float*)d_in[5];
    const float* b1  = (const float*)d_in[6];
    const float* W2  = (const float*)d_in[7];
    const float* b2  = (const float*)d_in[8];
    const float* Wg  = (const float*)d_in[9];
    const float* bg  = (const float*)d_in[10];
    const float* Wd  = (const float*)d_in[11];
    const float* bd  = (const float*)d_in[12];
    const float* Wt  = (const float*)d_in[13];
    const float* bt  = (const float*)d_in[14];
    float* out = (float*)d_out;

    _Float16* ws = (_Float16*)d_ws;
    _Float16* w1s = ws + WS_W1;
    _Float16* w2s = ws + WS_W2;
    _Float16* whs = ws + WS_WH;

    // weight swizzle+convert (tiny; weights stay L2 resident afterwards)
    delan_swz_w256<<<(65536 + 255) / 256, 256, 0, stream>>>(W1, w1s, 65536);
    delan_swz_w256<<<(65536 + 255) / 256, 256, 0, stream>>>(W2, w2s, 65536);
    delan_swz_heads<<<(12288 + 255) / 256, 256, 0, stream>>>(Wg, Wd, Wt, whs, 12288);

    const int smem = WAVES * 2 * ABUF_H * 2 + SPB * 8 * HB_COLS * 4; // 77824 B
    (void)hipFuncSetAttribute((const void*)delan_main,
                              hipFuncAttributeMaxDynamicSharedMemorySize, smem);
    delan_main<<<BATCH / SPB, WAVES * 32, smem, stream>>>(
        q, qd, qdd, W0, b0, b1, b2, bg, bd, bt, w1s, w2s, whs, out);
}
